// LearnOnlyGaussianPoolingV2_67834713473721
// MI455X (gfx1250) — matrix operand order repaired and checked by reading, not compiled, not run on hardware
//
#include <hip/hip_runtime.h>
#include <hip/hip_bf16.h>
#include <math.h>

typedef __attribute__((ext_vector_type(16))) _Float16 v16h;
typedef __attribute__((ext_vector_type(8)))  float    v8f;
typedef __attribute__((ext_vector_type(4)))  unsigned int v4u;
typedef __attribute__((ext_vector_type(8)))  int      v8i;
typedef __attribute__((ext_vector_type(4)))  int      v4i;

// Problem constants (fixed by the reference setup_inputs()).
constexpr int B = 4;
constexpr int N = 8192;
constexpr int C = 128;
constexpr int K = 683;
constexpr int KT = 43;          // ceil(K/16)
constexpr int KP = KT * 16;     // 688 padded slots

// d_out layout (floats): A | s_c | mu_c | Sigma_c | total
constexpr size_t OFF_SC = (size_t)B * N * K;
constexpr size_t OFF_MU = OFF_SC + (size_t)B * K * C;
constexpr size_t OFF_SG = OFF_MU + (size_t)B * K * 3;
constexpr size_t OFF_T  = OFF_SG + (size_t)B * K * 9;

// ws layout (bytes)
constexpr size_t WS_SPROJ = 0;                                   // B*N*C f16
constexpr size_t WS_ANCH  = 8388608;                             // B*K*3 f32
constexpr size_t WS_RED   = WS_ANCH + 33024;                     // B*K*24 f32
constexpr size_t WS_NORM  = WS_RED + (size_t)B * K * 24 * 4;     // B*K f32
constexpr size_t WS_SCAL  = WS_NORM + 11008;                     // 2 f32
constexpr size_t WS_SLOT  = WS_SCAL + 256;                       // K*C f16

// ---------------------------------------------------------------------------
// WMMA fragment helpers (CDNA5 16-bit A 16x32, B 32x16, f32 C/D 16x16 layouts)
// ---------------------------------------------------------------------------
__device__ __forceinline__ int a_kidx(int i, int half) {
  return i + half * 16;
}

// ---------------------------------------------------------------------------
// Tensor Data Mover: async 2D tile load (rows x cols f16) from global -> LDS.
// D# bitfields per CDNA5 ISA.
// ---------------------------------------------------------------------------
__device__ __forceinline__ void tdm_load_tile_f16(const _Float16* gsrc,
                                                  unsigned lds_off,
                                                  int tile_rows, int tile_cols,
                                                  int row_stride_elems,
                                                  int tensor_rows) {
  unsigned long long ga = (unsigned long long)(const void*)gsrc;
  v4u g0;
  g0[0] = 1u;                                            // count=1 (valid), user mode
  g0[1] = lds_off;                                       // lds_addr (bytes)
  g0[2] = (unsigned)(ga & 0xffffffffu);                  // global_addr[31:0]
  g0[3] = (unsigned)((ga >> 32) & 0x01ffffffu) | (2u << 30); // addr[56:32] | type=2
  v8i g1;
  g1[0] = (1 << 16);                                     // data_size=1 -> 2 bytes
  g1[1] = (int)((tile_cols & 0xffff) << 16);             // tensor_dim0[15:0]
  g1[2] = (int)(((tile_cols >> 16) & 0xffff) |           // tensor_dim0[31:16]
                ((tensor_rows & 0xffff) << 16));         // tensor_dim1[15:0]
  g1[3] = (int)((((unsigned)tensor_rows >> 16) & 0xffff) |
                ((tile_cols & 0xffff) << 16));           // tensor_dim1 hi | tile_dim0
  g1[4] = (int)(tile_rows & 0xffff);                     // tile_dim1 (tile_dim2=0)
  g1[5] = row_stride_elems;                              // tensor_dim0_stride[31:0]
  g1[6] = 0;
  g1[7] = 0;
  v4i gz = {0, 0, 0, 0};
#if __clang_major__ >= 23
  v8i gz8 = {0, 0, 0, 0, 0, 0, 0, 0};
  __builtin_amdgcn_tensor_load_to_lds(g0, g1, gz, gz, gz8, 0);
#else
  __builtin_amdgcn_tensor_load_to_lds(g0, g1, gz, gz, 0);
#endif
}

// ---------------------------------------------------------------------------
// Kernel 0: convert slot_embed f32 -> f16 once.
// ---------------------------------------------------------------------------
__global__ void k_cvt_slot(const float* __restrict__ slot, _Float16* __restrict__ slot16) {
  int i = blockIdx.x * blockDim.x + threadIdx.x;
  if (i < K * C) slot16[i] = (_Float16)slot[i];
}

// ---------------------------------------------------------------------------
// Kernel 1: farthest-point sampling, one workgroup per batch, md2 in LDS.
// ---------------------------------------------------------------------------
__global__ void k_fps(const float* __restrict__ mu, const float* __restrict__ mask,
                      float* __restrict__ anchors) {
  const int b = blockIdx.x;
  const float* muB = mu + (size_t)b * N * 3;
  const float* mB  = mask + (size_t)b * N;
  __shared__ float md2[N];           // 32 KB
  __shared__ float wv[32];
  __shared__ int   wi[32];
  __shared__ float bc[4];
  __shared__ float selmu[3];
  const int tid = threadIdx.x;       // 1024 threads
  const int lane = tid & 31, wave = tid >> 5;

  // masked mean of mu
  float sx = 0.f, sy = 0.f, sz = 0.f, c0 = 0.f;
  for (int n = tid; n < N; n += 1024) {
    float w = (mB[n] > 0.5f) ? 1.f : 0.f;
    sx += w * muB[n * 3 + 0];
    sy += w * muB[n * 3 + 1];
    sz += w * muB[n * 3 + 2];
    c0 += w;
  }
  for (int off = 16; off > 0; off >>= 1) {
    sx += __shfl_down(sx, off, 32);
    sy += __shfl_down(sy, off, 32);
    sz += __shfl_down(sz, off, 32);
    c0 += __shfl_down(c0, off, 32);
  }
  __shared__ float mred[32][4];
  if (lane == 0) { mred[wave][0] = sx; mred[wave][1] = sy; mred[wave][2] = sz; mred[wave][3] = c0; }
  __syncthreads();
  if (tid == 0) {
    float tx = 0, ty = 0, tz = 0, tc = 0;
    for (int w = 0; w < 32; ++w) { tx += mred[w][0]; ty += mred[w][1]; tz += mred[w][2]; tc += mred[w][3]; }
    tc = fmaxf(tc, 1.f);
    bc[0] = tx / tc; bc[1] = ty / tc; bc[2] = tz / tc;
  }
  __syncthreads();
  const float mx = bc[0], my = bc[1], mz = bc[2];

  // first = argmax masked distance-to-mean
  float bv = -3.4e38f; int bi = 0;
  for (int n = tid; n < N; n += 1024) {
    float dx = muB[n * 3] - mx, dy = muB[n * 3 + 1] - my, dz = muB[n * 3 + 2] - mz;
    float d2 = dx * dx + dy * dy + dz * dz;
    float v = (mB[n] > 0.5f) ? d2 : -1.f;
    if (v > bv) { bv = v; bi = n; }
  }
  for (int off = 16; off > 0; off >>= 1) {
    float ov = __shfl_down(bv, off, 32);
    int   oi = __shfl_down(bi, off, 32);
    if (ov > bv || (ov == bv && oi < bi)) { bv = ov; bi = oi; }
  }
  if (lane == 0) { wv[wave] = bv; wi[wave] = bi; }
  __syncthreads();
  if (tid == 0) {
    float v = wv[0]; int i = wi[0];
    for (int w = 1; w < 32; ++w)
      if (wv[w] > v || (wv[w] == v && wi[w] < i)) { v = wv[w]; i = wi[w]; }
    selmu[0] = muB[i * 3]; selmu[1] = muB[i * 3 + 1]; selmu[2] = muB[i * 3 + 2];
    anchors[((size_t)b * K + 0) * 3 + 0] = selmu[0];
    anchors[((size_t)b * K + 0) * 3 + 1] = selmu[1];
    anchors[((size_t)b * K + 0) * 3 + 2] = selmu[2];
  }
  __syncthreads();
  {
    float ax = selmu[0], ay = selmu[1], az = selmu[2];
    for (int n = tid; n < N; n += 1024) {
      float dx = muB[n * 3] - ax, dy = muB[n * 3 + 1] - ay, dz = muB[n * 3 + 2] - az;
      float d2 = dx * dx + dy * dy + dz * dz;
      md2[n] = (mB[n] > 0.5f) ? d2 : -1.f;
    }
  }
  __syncthreads();

  for (int k = 1; k < K; ++k) {
    float v = -3.4e38f; int i = 0;
    for (int n = tid; n < N; n += 1024) {
      float m = md2[n];
      if (m > v) { v = m; i = n; }
    }
    for (int off = 16; off > 0; off >>= 1) {
      float ov = __shfl_down(v, off, 32);
      int   oi = __shfl_down(i, off, 32);
      if (ov > v || (ov == v && oi < i)) { v = ov; i = oi; }
    }
    if (lane == 0) { wv[wave] = v; wi[wave] = i; }
    __syncthreads();
    if (tid == 0) {
      float fv = wv[0]; int fi = wi[0];
      for (int w = 1; w < 32; ++w)
        if (wv[w] > fv || (wv[w] == fv && wi[w] < fi)) { fv = wv[w]; fi = wi[w]; }
      selmu[0] = muB[fi * 3]; selmu[1] = muB[fi * 3 + 1]; selmu[2] = muB[fi * 3 + 2];
      anchors[((size_t)b * K + k) * 3 + 0] = selmu[0];
      anchors[((size_t)b * K + k) * 3 + 1] = selmu[1];
      anchors[((size_t)b * K + k) * 3 + 2] = selmu[2];
    }
    __syncthreads();
    float ax = selmu[0], ay = selmu[1], az = selmu[2];
    for (int n = tid; n < N; n += 1024) {
      float dx = muB[n * 3] - ax, dy = muB[n * 3 + 1] - ay, dz = muB[n * 3 + 2] - az;
      float d2 = dx * dx + dy * dy + dz * dz;
      float nd = (mB[n] > 0.5f) ? d2 : -1.f;
      float om = md2[n];
      md2[n] = (om == -1.f) ? nd : fminf(om, nd);
    }
    __syncthreads();
  }
}

// ---------------------------------------------------------------------------
// Kernel 2: s_proj = s @ W_proj^T via WMMA, f16 output.
// ---------------------------------------------------------------------------
__global__ void k_sproj(const float* __restrict__ s, const float* __restrict__ Wp,
                        _Float16* __restrict__ sproj) {
  int lin = blockIdx.x;
  const int ct = lin & 7;
  const int mt = lin >> 3;
  const int lane = threadIdx.x & 31;
  const int r = lane & 15, half = lane >> 4;
  const float* arow = s + (size_t)mt * 16 * C;
  v8f acc = {};
  for (int kk = 0; kk < C; kk += 32) {
    v16h af, bf;
#pragma unroll
    for (int i = 0; i < 16; ++i)
      af[i] = (_Float16)arow[r * C + kk + a_kidx(i, half)];
    const int dcol = ct * 16 + r;
#pragma unroll
    for (int i = 0; i < 16; ++i)
      bf[i] = (_Float16)Wp[dcol * C + kk + i + half * 16];
    acc = __builtin_amdgcn_wmma_f32_16x16x32_f16(false, af, false, bf,
                                                 (short)0, acc, false, false);
  }
  const int col = ct * 16 + (lane & 15);
#pragma unroll
  for (int i = 0; i < 8; ++i) {
    const int row = mt * 16 + i + 8 * half;
    sproj[(size_t)row * C + col] = (_Float16)acc[i];
  }
}

// ---------------------------------------------------------------------------
// Kernel 3: fused logits + masked softmax.
// A-tile (16x128 f16) staged into LDS once per block via the TDM.
// ---------------------------------------------------------------------------
__global__ void k_assign(const _Float16* __restrict__ sproj,
                         const _Float16* __restrict__ slot16,
                         const float* __restrict__ mu,
                         const float* __restrict__ mask,
                         const float* __restrict__ anchors,
                         const float* __restrict__ lsm,
                         const float* __restrict__ wsem_p,
                         const float* __restrict__ wmu_p,
                         float* __restrict__ A) {
  const int blk = blockIdx.x;
  const int b = blk / (N / 16);
  const int m0 = (blk % (N / 16)) * 16;
  __shared__ float lg[16][KP];          // 44 KB
  __shared__ _Float16 atile[16 * C];    // 4 KB, filled by TDM
  __shared__ float muR[16][3];
  __shared__ float mrow[16];
  __shared__ float red[16][16];
  __shared__ float rmax[16], rsum[16];
  const int tid = threadIdx.x;          // 256
  const int wave = tid >> 5, lane = tid & 31;

  // TDM: async-load the s_proj tile for these 16 rows into LDS (one wave issues).
  if (wave == 0) {
    tdm_load_tile_f16(sproj + ((size_t)b * N + m0) * C,
                      (unsigned)(unsigned long long)(const void*)&atile[0],
                      /*rows*/16, /*cols*/C, /*stride*/C, /*tensor rows*/N);
    __builtin_amdgcn_s_wait_tensorcnt(0);
  }
  if (tid < 16) mrow[tid] = mask[(size_t)b * N + m0 + tid];
  if (tid < 48) muR[tid / 3][tid % 3] = mu[((size_t)b * N + m0 + tid / 3) * 3 + tid % 3];
  const float sig = fmaxf(expf(lsm[0]), 1e-4f);
  const float inv2s2 = 1.f / (2.f * sig * sig);
  const float wsem = wsem_p[0], wmu = wmu_p[0];
  __syncthreads();

  const int r = lane & 15, half = lane >> 4;
  for (int ct = wave; ct < KT; ct += 8) {
    // prefetch next slot tile for this wave
    if (ct + 8 < KT)
      __builtin_prefetch(slot16 + (size_t)(ct + 8) * 16 * C, 0, 0);
    v8f acc = {};
    for (int kk = 0; kk < C; kk += 32) {
      v16h af, bf;
#pragma unroll
      for (int i = 0; i < 16; ++i)
        af[i] = atile[r * C + kk + a_kidx(i, half)];
      const int kcol = ct * 16 + r;
#pragma unroll
      for (int i = 0; i < 16; ++i)
        bf[i] = (kcol < K) ? slot16[(size_t)kcol * C + kk + i + half * 16]
                           : (_Float16)0.f;
      acc = __builtin_amdgcn_wmma_f32_16x16x32_f16(false, af, false, bf,
                                                   (short)0, acc, false, false);
    }
    const int col = lane & 15;
    const int kcol = ct * 16 + col;
    float ax = 0.f, ay = 0.f, az = 0.f;
    if (kcol < K) {
      ax = anchors[((size_t)b * K + kcol) * 3 + 0];
      ay = anchors[((size_t)b * K + kcol) * 3 + 1];
      az = anchors[((size_t)b * K + kcol) * 3 + 2];
    }
#pragma unroll
    for (int i = 0; i < 8; ++i) {
      const int row = i + 8 * half;
      float v;
      if (kcol >= K) {
        v = -1e30f;
      } else {
        float dx = muR[row][0] - ax, dy = muR[row][1] - ay, dz = muR[row][2] - az;
        float d2 = dx * dx + dy * dy + dz * dz;
        v = wsem * acc[i] + wmu * (-d2 * inv2s2);   // TAU = 1
        if (mrow[row] < 0.5f) v = -1e9f;
      }
      lg[row][ct * 16 + col] = v;
    }
  }
  __syncthreads();

  const int row = tid >> 4, p = tid & 15;
  float m = -3.4e38f;
  for (int c = p; c < KP; c += 16) m = fmaxf(m, lg[row][c]);
  red[row][p] = m;
  __syncthreads();
  if (p == 0) {
    float mm = red[row][0];
    for (int j = 1; j < 16; ++j) mm = fmaxf(mm, red[row][j]);
    rmax[row] = mm;
  }
  __syncthreads();
  float s0 = 0.f;
  for (int c = p; c < KP; c += 16) s0 += expf(lg[row][c] - rmax[row]);
  red[row][p] = s0;
  __syncthreads();
  if (p == 0) {
    float ss = 0.f;
    for (int j = 0; j < 16; ++j) ss += red[row][j];
    rsum[row] = ss;
  }
  __syncthreads();
  const float inv = 1.f / rsum[row];
  const float mr = mrow[row];
  for (int c = p; c < K; c += 16)
    A[((size_t)b * N + m0 + row) * K + c] = expf(lg[row][c] - rmax[row]) * inv * mr;
}

// ---------------------------------------------------------------------------
// Kernel 4: per-(b,k) moments.
// ---------------------------------------------------------------------------
__global__ void k_red(const float* __restrict__ A, const float* __restrict__ mu,
                      const float* __restrict__ Sigma, float* __restrict__ red) {
  const int bid = blockIdx.x;
  const int b = bid / K, k = bid % K;
  const int tid = threadIdx.x;   // 256
  float acc[23];
#pragma unroll
  for (int t = 0; t < 23; ++t) acc[t] = 0.f;
  for (int n = tid; n < N; n += 256) {
    const float a = A[((size_t)b * N + n) * K + k];
    const float mx = mu[((size_t)b * N + n) * 3 + 0];
    const float my = mu[((size_t)b * N + n) * 3 + 1];
    const float mz = mu[((size_t)b * N + n) * 3 + 2];
    acc[0] += a; acc[1] += a * a;
    acc[2] += a * mx; acc[3] += a * my; acc[4] += a * mz;
    const float* S = Sigma + ((size_t)b * N + n) * 9;
#pragma unroll
    for (int j = 0; j < 9; ++j) acc[5 + j] += a * S[j];
    acc[14] += a * mx * mx; acc[15] += a * mx * my; acc[16] += a * mx * mz;
    acc[17] += a * my * mx; acc[18] += a * my * my; acc[19] += a * my * mz;
    acc[20] += a * mz * mx; acc[21] += a * mz * my; acc[22] += a * mz * mz;
  }
#pragma unroll
  for (int t = 0; t < 23; ++t)
    for (int off = 16; off > 0; off >>= 1) acc[t] += __shfl_down(acc[t], off, 32);
  __shared__ float buf[8][23];
  const int wave = tid >> 5, lane = tid & 31;
  if (lane == 0)
    for (int t = 0; t < 23; ++t) buf[wave][t] = acc[t];
  __syncthreads();
  if (tid == 0) {
    for (int t = 0; t < 23; ++t) {
      float x = 0.f;
      for (int w = 0; w < 8; ++w) x += buf[w][t];
      red[(size_t)bid * 24 + t] = x;
    }
  }
}

// ---------------------------------------------------------------------------
// Kernel 5: s_c numerator = A^T @ s via WMMA.
// ---------------------------------------------------------------------------
__global__ void k_sc(const float* __restrict__ A, const float* __restrict__ s,
                     float* __restrict__ sc) {
  int lin = blockIdx.x;
  const int ct = lin % 8; lin /= 8;
  const int kt = lin % KT; const int b = lin / KT;
  const int lane = threadIdx.x & 31;
  const int r = lane & 15, half = lane >> 4;
  const float* Ab = A + (size_t)b * N * K;
  const float* sb = s + (size_t)b * N * C;
  const int kg = kt * 16 + r;
  v8f acc = {};
  for (int n0 = 0; n0 < N; n0 += 32) {
    v16h af, bf;
#pragma unroll
    for (int i = 0; i < 16; ++i) {
      const int nq = n0 + a_kidx(i, half);
      af[i] = (kg < K) ? (_Float16)Ab[(size_t)nq * K + kg] : (_Float16)0.f;
    }
    const int cg = ct * 16 + r;
#pragma unroll
    for (int i = 0; i < 16; ++i) {
      const int nq = n0 + i + half * 16;
      bf[i] = (_Float16)sb[(size_t)nq * C + cg];
    }
    acc = __builtin_amdgcn_wmma_f32_16x16x32_f16(false, af, false, bf,
                                                 (short)0, acc, false, false);
  }
  const int cg = ct * 16 + (lane & 15);
#pragma unroll
  for (int i = 0; i < 8; ++i) {
    const int kgo = kt * 16 + i + 8 * half;
    if (kgo < K) sc[((size_t)b * K + kgo) * C + cg] = acc[i];
  }
}

// ---------------------------------------------------------------------------
// Kernel 6: finalize per (b,k).
// ---------------------------------------------------------------------------
__global__ void k_fin(const float* __restrict__ red, float* __restrict__ sc,
                      float* __restrict__ mu_c, float* __restrict__ Sigma_c,
                      float* __restrict__ norms) {
  const int bid = blockIdx.x;
  const float* r = red + (size_t)bid * 24;
  const float denom = fmaxf(r[0], 1e-8f);
  const int tid = threadIdx.x;   // 128
  sc[(size_t)bid * C + tid] /= denom;
  if (tid < 3) mu_c[(size_t)bid * 3 + tid] = r[2 + tid] / denom;
  if (tid == 0) norms[bid] = fmaxf(sqrtf(r[1]), 1e-8f);
  if (tid < 9) {
    const int i = tid / 3, j = tid % 3;
    const float Ssum = r[0] / denom;
    const float mci = r[2 + i] / denom, mcj = r[2 + j] / denom;
    const float intra_ij = r[5 + i * 3 + j] / denom;
    const float intra_ji = r[5 + j * 3 + i] / denom;
    const float M2_ij = r[14 + i * 3 + j] / denom;
    const float M2_ji = r[14 + j * 3 + i] / denom;
    const float inter_ij = M2_ij - 2.f * mci * mcj + Ssum * mci * mcj;
    const float inter_ji = M2_ji - 2.f * mcj * mci + Ssum * mcj * mci;
    float v = 0.5f * ((intra_ij + inter_ij) + (intra_ji + inter_ji));
    if (i == j) v += 1e-6f;
    Sigma_c[(size_t)bid * 9 + tid] = v;
  }
}

// ---------------------------------------------------------------------------
// Kernel 7: occupancy loss -> scal[0]
// ---------------------------------------------------------------------------
__global__ void k_occ(const float* __restrict__ red, float* __restrict__ scal) {
  __shared__ float sb[256];
  const int tid = threadIdx.x;
  float accum = 0.f;
  for (int b = 0; b < B; ++b) {
    float ps = 0.f;
    for (int k = tid; k < K; k += 256) ps += red[((size_t)b * K + k) * 24];
    sb[tid] = ps; __syncthreads();
    for (int off = 128; off > 0; off >>= 1) {
      if (tid < off) sb[tid] += sb[tid + off];
      __syncthreads();
    }
    const float tot = fmaxf(sb[0], 1e-8f);
    __syncthreads();
    float psq = 0.f;
    const float invK = 1.f / (float)K;
    for (int k = tid; k < K; k += 256) {
      const float occ = red[((size_t)b * K + k) * 24] / tot;
      const float d = occ - invK;
      psq += d * d;
    }
    sb[tid] = psq; __syncthreads();
    for (int off = 128; off > 0; off >>= 1) {
      if (tid < off) sb[tid] += sb[tid + off];
      __syncthreads();
    }
    if (tid == 0) accum += sb[0];
    __syncthreads();
  }
  if (tid == 0) scal[0] = accum / (float)(B * K);
}

// ---------------------------------------------------------------------------
// Kernel 8: collapse Gram via WMMA; sum of squared off-diagonals.
// ---------------------------------------------------------------------------
__global__ void k_gram(const float* __restrict__ A, const float* __restrict__ norms,
                       float* __restrict__ acc_out) {
  int lin = blockIdx.x;
  const int lt = lin % KT; lin /= KT;
  const int kt = lin % KT; const int b = lin / KT;
  const int lane = threadIdx.x & 31;
  const int r = lane & 15, half = lane >> 4;
  const float* Ab = A + (size_t)b * N * K;
  const int kg = kt * 16 + r;
  const int lgc = lt * 16 + r;
  v8f acc = {};
  for (int n0 = 0; n0 < N; n0 += 32) {
    v16h af, bf;
#pragma unroll
    for (int i = 0; i < 16; ++i) {
      const int nq = n0 + a_kidx(i, half);
      af[i] = (kg < K) ? (_Float16)Ab[(size_t)nq * K + kg] : (_Float16)0.f;
    }
#pragma unroll
    for (int i = 0; i < 16; ++i) {
      const int nq = n0 + i + half * 16;
      bf[i] = (lgc < K) ? (_Float16)Ab[(size_t)nq * K + lgc] : (_Float16)0.f;
    }
    acc = __builtin_amdgcn_wmma_f32_16x16x32_f16(false, af, false, bf,
                                                 (short)0, acc, false, false);
  }
  const float* nb = norms + (size_t)b * K;
  const int gl = lt * 16 + (lane & 15);
  float ssq = 0.f;
#pragma unroll
  for (int i = 0; i < 8; ++i) {
    const int gk = kt * 16 + i + 8 * half;
    if (gk < K && gl < K && gk != gl) {
      const float g = acc[i] / (nb[gk] * nb[gl]);
      ssq += g * g;
    }
  }
  for (int off = 16; off > 0; off >>= 1) ssq += __shfl_down(ssq, off, 32);
  if (lane == 0) atomicAdd(acc_out, ssq);
}

// ---------------------------------------------------------------------------
// Kernel 9: total loss.
// ---------------------------------------------------------------------------
__global__ void k_total(const float* __restrict__ scal, float* __restrict__ out_total) {
  if (threadIdx.x == 0)
    out_total[0] = 1.0f * scal[0] + 0.1f * (scal[1] / ((float)B * K * K));
}

// ---------------------------------------------------------------------------
extern "C" void kernel_launch(void* const* d_in, const int* in_sizes, int n_in,
                              void* d_out, int out_size, void* d_ws, size_t ws_size,
                              hipStream_t stream) {
  (void)in_sizes; (void)n_in; (void)out_size; (void)ws_size;
  const float* s     = (const float*)d_in[0];
  const float* mu    = (const float*)d_in[1];
  const float* Sigma = (const float*)d_in[2];
  const float* mask  = (const float*)d_in[3];
  const float* Wp    = (const float*)d_in[4];
  const float* slot  = (const float*)d_in[5];
  const float* lsm   = (const float*)d_in[6];
  const float* wsem  = (const float*)d_in[7];
  const float* wmu   = (const float*)d_in[8];

  float* out = (float*)d_out;
  float* A       = out;
  float* sc      = out + OFF_SC;
  float* mu_c    = out + OFF_MU;
  float* Sigma_c = out + OFF_SG;
  float* total   = out + OFF_T;

  char* ws = (char*)d_ws;
  _Float16* sproj  = (_Float16*)(ws + WS_SPROJ);
  float* anchors   = (float*)(ws + WS_ANCH);
  float* red       = (float*)(ws + WS_RED);
  float* norms     = (float*)(ws + WS_NORM);
  float* scal      = (float*)(ws + WS_SCAL);
  _Float16* slot16 = (_Float16*)(ws + WS_SLOT);

  hipMemsetAsync(scal, 0, 2 * sizeof(float), stream);

  k_cvt_slot<<<(K * C + 255) / 256, 256, 0, stream>>>(slot, slot16);
  k_fps   <<<B, 1024, 0, stream>>>(mu, mask, anchors);
  k_sproj <<<(B * N / 16) * (C / 16), 32, 0, stream>>>(s, Wp, sproj);
  k_assign<<<B * (N / 16), 256, 0, stream>>>(sproj, slot16, mu, mask, anchors,
                                             lsm, wsem, wmu, A);
  k_red   <<<B * K, 256, 0, stream>>>(A, mu, Sigma, red);
  k_sc    <<<B * KT * (C / 16), 32, 0, stream>>>(A, s, sc);
  k_fin   <<<B * K, 128, 0, stream>>>(red, sc, mu_c, Sigma_c, norms);
  k_occ   <<<1, 256, 0, stream>>>(red, scal);
  k_gram  <<<B * KT * KT, 32, 0, stream>>>(A, norms, scal + 1);
  k_total <<<1, 1, 0, stream>>>(scal, total);
}
